// se_equivariant_net_47991964565505
// MI455X (gfx1250) — matrix operand order repaired and checked by reading, not compiled, not run on hardware
//
#include <hip/hip_runtime.h>
#include <math.h>

#define LAYERS 3
#define BB 4
#define NN 384
#define DD 3
#define HD 32
#define MLPW 64
#define EIN 65
#define ITILES (NN / 16)  // 24

typedef __attribute__((ext_vector_type(16))) _Float16 v16h;
typedef __attribute__((ext_vector_type(8)))  float    v8f;

static __device__ __forceinline__ v8f wmma_f16(v16h a, v16h b, v8f c) {
  return __builtin_amdgcn_wmma_f32_16x16x32_f16(false, a, false, b, (short)0, c, false, false);
}

// B fragment: 32(K) x 16(N) f16. Lane l holds column n0+l%16; elems 0..7 = K kc+hs*8+0..7, elems 8..15 = K kc+16+hs*8+0..7
static __device__ __forceinline__ v16h load_bfrag(const _Float16* W, int n0, int kc, int lane) {
  const _Float16* p = W + kc * MLPW + n0 + (lane & 15);
  int hs = lane >> 4;
  v16h b;
#pragma unroll
  for (int k = 0; k < 8; ++k) {
    b[k]     = p[(hs * 8 + k) * MLPW];
    b[k + 8] = p[(16 + hs * 8 + k) * MLPW];
  }
  return b;
}

// A fragment: 16(M) x 32(K) f16 from an f32 row in LDS.
// lane l (row = l%16, hs = l/16): elems 0..7 = K kc+hs*8+0..7, elems 8..15 = K kc+16+hs*8+0..7
static __device__ __forceinline__ v16h pack_afrag(const float* row, int kc, int hs) {
  v16h a;
#pragma unroll
  for (int k = 0; k < 8; ++k) {
    a[k]     = (_Float16)row[kc + hs * 8 + k];
    a[k + 8] = (_Float16)row[kc + 16 + hs * 8 + k];
  }
  return a;
}

__global__ void k_init(const float* __restrict__ x, float* x_cur, float* h_cur) {
  int i = blockIdx.x * blockDim.x + threadIdx.x;
  if (i < BB * NN * DD) x_cur[i] = x[i];
  if (i < BB * NN * HD) h_cur[i] = 0.f;
}

__global__ void k_cvtw(const float* __restrict__ pe_w1, const float* __restrict__ px_w0,
                       const float* __restrict__ px_w1, _Float16* __restrict__ w16) {
  int i = blockIdx.x * blockDim.x + threadIdx.x;
  if (i >= LAYERS * 3 * 4096) return;
  int l = i / (3 * 4096);
  int rem = i % (3 * 4096);
  int m = rem / 4096, kn = rem % 4096;
  const float* src = (m == 0 ? pe_w1 : (m == 1 ? px_w0 : px_w1));
  w16[i] = (_Float16)src[l * 4096 + kn];
}

// Per-node first-edge-layer precompute: Hj = h @ W[1:33], HiB = h @ W[33:65] + b0
__global__ void k_prep(const float* __restrict__ h_cur, const float* __restrict__ pe_w0,
                       const float* __restrict__ pe_b0, float* __restrict__ Hj,
                       float* __restrict__ HiB, int l) {
  int node = blockIdx.x;       // b*N + n
  int f = threadIdx.x;         // 0..63
  const float* hrow = h_cur + node * HD;
  const float* W = pe_w0 + l * EIN * MLPW;
  float aj = 0.f, ai = 0.f;
  for (int k = 0; k < HD; ++k) {
    float hv = hrow[k];
    aj += hv * W[(1 + k) * MLPW + f];
    ai += hv * W[(1 + HD + k) * MLPW + f];
  }
  Hj[node * MLPW + f] = aj;
  HiB[node * MLPW + f] = ai + pe_b0[l * MLPW + f];
}

__global__ __launch_bounds__(128) void k_edge(
    const float* __restrict__ x_cur, const float* __restrict__ Hj,
    const float* __restrict__ HiB, const float* __restrict__ pe_w0,
    const float* __restrict__ pe_b1, const float* __restrict__ inf_w,
    const float* __restrict__ inf_b, const float* __restrict__ px_b0,
    const float* __restrict__ px_b1, const float* __restrict__ px_w2,
    const float* __restrict__ px_b2, const _Float16* __restrict__ w16,
    float* __restrict__ m_i, float* __restrict__ shiftv, int l) {
  __shared__ _Float16 wA[3 * 4096];       // pe_w1, px_w0, px_w1 (f16, row-major 64x64)
  __shared__ float slab[4][16 * 64];      // per-wave activation bounce / final m partials
  __shared__ float e_s[4][16];
  __shared__ float sp[4][48];
  __shared__ float s_infw[64], s_pxw2[64];

  int tid = threadIdx.x;
  const _Float16* wsrc = w16 + l * 3 * 4096;
  for (int i = tid; i < 3 * 4096; i += 128) wA[i] = wsrc[i];
  if (tid < 64) {
    s_infw[tid] = inf_w[l * 64 + tid];
    s_pxw2[tid] = px_w2[l * 64 + tid];
  }
  __syncthreads();

  int wid = tid >> 5, lane = tid & 31;
  int r = lane & 15, hs = lane >> 4;
  int b = blockIdx.x / ITILES;
  int i0 = (blockIdx.x % ITILES) * 16;
  int nodei = b * NN + i0 + r;

  float xi0 = x_cur[nodei * 3 + 0];
  float xi1 = x_cur[nodei * 3 + 1];
  float xi2 = x_cur[nodei * 3 + 2];

  // loop-invariant per-lane: HiB and pe_w0 row 0 for this lane's 32 A-fragment features
  float hib[32], w0c[32];
  const float* w0row = pe_w0 + l * EIN * MLPW;  // row 0 (sq column)
#pragma unroll
  for (int c = 0; c < 2; ++c) {
#pragma unroll
    for (int k = 0; k < 8; ++k) {
      int f1 = c * 32 + hs * 8 + k;
      int f2 = c * 32 + 16 + hs * 8 + k;
      hib[c * 16 + k]     = HiB[nodei * 64 + f1];
      hib[c * 16 + 8 + k] = HiB[nodei * 64 + f2];
      w0c[c * 16 + k]     = w0row[f1];
      w0c[c * 16 + 8 + k] = w0row[f2];
    }
  }

  float bpe[4], bp0[4], bp1[4];
#pragma unroll
  for (int t = 0; t < 4; ++t) {
    bpe[t] = pe_b1[l * 64 + t * 16 + r];
    bp0[t] = px_b0[l * 64 + t * 16 + r];
    bp1[t] = px_b1[l * 64 + t * 16 + r];
  }
  float infb = inf_b[l];
  float pxb2 = px_b2[l];

  v8f macc[4];
#pragma unroll
  for (int t = 0; t < 4; ++t) {
#pragma unroll
    for (int g = 0; g < 8; ++g) macc[t][g] = 0.f;
  }
  float sh0 = 0.f, sh1 = 0.f, sh2 = 0.f;
  float* myslab = slab[wid];

  for (int j = wid; j < NN; j += 4) {
    int nodej = b * NN + j;
    float xj0 = x_cur[nodej * 3 + 0];
    float xj1 = x_cur[nodej * 3 + 1];
    float xj2 = x_cur[nodej * 3 + 2];
    float d0 = xj0 - xi0, d1 = xj1 - xi1, d2 = xj2 - xi2;
    float sq = d0 * d0 + d1 * d1 + d2 * d2;
    bool diag = ((i0 + r) == j);
    const float* hjrow = Hj + nodej * 64;

    // ---- phi_e layer 2: m = relu(m1) @ pe_w1 + b1, m1 built directly in A layout
    v8f mc[4];
#pragma unroll
    for (int t = 0; t < 4; ++t) {
#pragma unroll
      for (int g = 0; g < 8; ++g) mc[t][g] = bpe[t];
    }
#pragma unroll
    for (int c = 0; c < 2; ++c) {
      v16h a;
#pragma unroll
      for (int k = 0; k < 8; ++k) {
        int f1 = c * 32 + hs * 8 + k;
        int f2 = c * 32 + 16 + hs * 8 + k;
        float v1 = sq * w0c[c * 16 + k]     + hjrow[f1] + hib[c * 16 + k];
        float v2 = sq * w0c[c * 16 + 8 + k] + hjrow[f2] + hib[c * 16 + 8 + k];
        a[k]     = (_Float16)fmaxf(v1, 0.f);
        a[k + 8] = (_Float16)fmaxf(v2, 0.f);
      }
#pragma unroll
      for (int t = 0; t < 4; ++t)
        mc[t] = wmma_f16(a, load_bfrag(&wA[0], t * 16, c * 32, lane), mc[t]);
    }

    // store m (f32, C layout -> plain 16x64) into the wave slab
#pragma unroll
    for (int t = 0; t < 4; ++t) {
#pragma unroll
      for (int g = 0; g < 8; ++g)
        myslab[(g + 8 * hs) * 64 + t * 16 + r] = mc[t][g];
    }

    // ---- phi_inf: e = sigmoid(m . inf_w + inf_b), zero on diagonal
    if (hs == 0) {
      float acc = infb;
      for (int f = 0; f < 64; ++f) acc += myslab[r * 64 + f] * s_infw[f];
      e_s[wid][r] = diag ? 0.f : 1.f / (1.f + __expf(-acc));
    }
    float ev8[8];
#pragma unroll
    for (int g = 0; g < 8; ++g) ev8[g] = e_s[wid][g + 8 * hs];
#pragma unroll
    for (int t = 0; t < 4; ++t) {
#pragma unroll
      for (int g = 0; g < 8; ++g) macc[t][g] += mc[t][g] * ev8[g];
    }

    // ---- phi_x layer 1: relu(m @ px_w0 + b)
    v8f c1[4];
#pragma unroll
    for (int t = 0; t < 4; ++t) {
#pragma unroll
      for (int g = 0; g < 8; ++g) c1[t][g] = bp0[t];
    }
#pragma unroll
    for (int c = 0; c < 2; ++c) {
      v16h a = pack_afrag(&myslab[r * 64], c * 32, hs);
#pragma unroll
      for (int t = 0; t < 4; ++t)
        c1[t] = wmma_f16(a, load_bfrag(&wA[4096], t * 16, c * 32, lane), c1[t]);
    }
#pragma unroll
    for (int t = 0; t < 4; ++t) {
#pragma unroll
      for (int g = 0; g < 8; ++g)
        myslab[(g + 8 * hs) * 64 + t * 16 + r] = fmaxf(c1[t][g], 0.f);
    }

    // ---- phi_x layer 2: relu(. @ px_w1 + b)
    v8f c2[4];
#pragma unroll
    for (int t = 0; t < 4; ++t) {
#pragma unroll
      for (int g = 0; g < 8; ++g) c2[t][g] = bp1[t];
    }
#pragma unroll
    for (int c = 0; c < 2; ++c) {
      v16h a = pack_afrag(&myslab[r * 64], c * 32, hs);
#pragma unroll
      for (int t = 0; t < 4; ++t)
        c2[t] = wmma_f16(a, load_bfrag(&wA[2 * 4096], t * 16, c * 32, lane), c2[t]);
    }
#pragma unroll
    for (int t = 0; t < 4; ++t) {
#pragma unroll
      for (int g = 0; g < 8; ++g)
        myslab[(g + 8 * hs) * 64 + t * 16 + r] = fmaxf(c2[t][g], 0.f);
    }

    // ---- phi_x output + coordinate shift accumulation
    if (hs == 0) {
      float acc = pxb2;
      for (int f = 0; f < 64; ++f) acc += myslab[r * 64 + f] * s_pxw2[f];
      float pxo = diag ? 0.f : acc;
      float w = pxo / (sq + 1.f);
      sh0 += d0 * w;
      sh1 += d1 * w;
      sh2 += d2 * w;
    }
  }

  // dump per-wave partials, reduce across the 4 waves
#pragma unroll
  for (int t = 0; t < 4; ++t) {
#pragma unroll
    for (int g = 0; g < 8; ++g)
      myslab[(g + 8 * hs) * 64 + t * 16 + r] = macc[t][g];
  }
  if (hs == 0) {
    sp[wid][r * 3 + 0] = sh0;
    sp[wid][r * 3 + 1] = sh1;
    sp[wid][r * 3 + 2] = sh2;
  }
  __syncthreads();

  for (int idx = tid * 8; idx < tid * 8 + 8; ++idx) {
    float v = slab[0][idx] + slab[1][idx] + slab[2][idx] + slab[3][idx];
    int row = idx >> 6, col = idx & 63;
    m_i[(b * NN + i0 + row) * 64 + col] = v;
  }
  if (tid < 48) {
    float v = sp[0][tid] + sp[1][tid] + sp[2][tid] + sp[3][tid];
    shiftv[(b * NN + i0 + tid / 3) * 3 + tid % 3] = v / (float)(NN - 1);
  }
}

// phi_h node update + coordinate update
__global__ void k_node(const float* __restrict__ m_i, const float* __restrict__ shiftv,
                       const float* __restrict__ ph_w0, const float* __restrict__ ph_b0,
                       const float* __restrict__ ph_w1, const float* __restrict__ ph_b1,
                       const float* __restrict__ ph_w2, const float* __restrict__ ph_b2,
                       float* __restrict__ h_cur, float* __restrict__ x_cur, int l) {
  __shared__ float hin[96], a1[64], a2[64];
  int node = blockIdx.x;
  int t = threadIdx.x;  // 64 threads
  hin[t] = m_i[node * 64 + t];
  if (t < 32) hin[64 + t] = h_cur[node * 32 + t];
  __syncthreads();
  const float* W0 = ph_w0 + l * 96 * 64;
  float acc = ph_b0[l * 64 + t];
  for (int k = 0; k < 96; ++k) acc += hin[k] * W0[k * 64 + t];
  a1[t] = fmaxf(acc, 0.f);
  __syncthreads();
  const float* W1 = ph_w1 + l * 64 * 64;
  acc = ph_b1[l * 64 + t];
  for (int k = 0; k < 64; ++k) acc += a1[k] * W1[k * 64 + t];
  a2[t] = acc;
  __syncthreads();
  if (t < 32) {
    const float* W2 = ph_w2 + l * 64 * 32;
    float u = ph_b2[l * 32 + t];
    for (int k = 0; k < 64; ++k) u += a2[k] * W2[k * 32 + t];
    h_cur[node * 32 + t] += u;
  }
  if (t >= 32 && t < 35) {
    int d = t - 32;
    x_cur[node * 3 + d] += shiftv[node * 3 + d];
  }
}

__global__ void k_out(const float* __restrict__ x_cur, float* __restrict__ out) {
  int i = blockIdx.x * blockDim.x + threadIdx.x;
  if (i < BB * NN * DD) out[i] = x_cur[i];
}

extern "C" void kernel_launch(void* const* d_in, const int* in_sizes, int n_in,
                              void* d_out, int out_size, void* d_ws, size_t ws_size,
                              hipStream_t stream) {
  (void)in_sizes; (void)n_in; (void)out_size; (void)ws_size;
  const float* x     = (const float*)d_in[0];
  const float* pe_w0 = (const float*)d_in[1];
  const float* pe_b0 = (const float*)d_in[2];
  const float* pe_w1 = (const float*)d_in[3];
  const float* pe_b1 = (const float*)d_in[4];
  const float* inf_w = (const float*)d_in[5];
  const float* inf_b = (const float*)d_in[6];
  const float* px_w0 = (const float*)d_in[7];
  const float* px_b0 = (const float*)d_in[8];
  const float* px_w1 = (const float*)d_in[9];
  const float* px_b1 = (const float*)d_in[10];
  const float* px_w2 = (const float*)d_in[11];
  const float* px_b2 = (const float*)d_in[12];
  const float* ph_w0 = (const float*)d_in[13];
  const float* ph_b0 = (const float*)d_in[14];
  const float* ph_w1 = (const float*)d_in[15];
  const float* ph_b1 = (const float*)d_in[16];
  const float* ph_w2 = (const float*)d_in[17];
  const float* ph_b2 = (const float*)d_in[18];

  float* ws    = (float*)d_ws;
  float* x_cur = ws + 0;        // 4608
  float* h_cur = ws + 4608;     // 49152
  float* Hj    = ws + 53760;    // 98304
  float* HiB   = ws + 152064;   // 98304
  float* m_i   = ws + 250368;   // 98304
  float* shf   = ws + 348672;   // 4608
  _Float16* w16 = (_Float16*)(ws + 353280);  // 36864 halves

  k_init<<<192, 256, 0, stream>>>(x, x_cur, h_cur);
  k_cvtw<<<(LAYERS * 3 * 4096 + 255) / 256, 256, 0, stream>>>(pe_w1, px_w0, px_w1, w16);
  for (int l = 0; l < LAYERS; ++l) {
    k_prep<<<BB * NN, 64, 0, stream>>>(h_cur, pe_w0, pe_b0, Hj, HiB, l);
    k_edge<<<BB * ITILES, 128, 0, stream>>>(x_cur, Hj, HiB, pe_w0, pe_b1, inf_w, inf_b,
                                            px_b0, px_b1, px_w2, px_b2, w16, m_i, shf, l);
    k_node<<<BB * NN, 64, 0, stream>>>(m_i, shf, ph_w0, ph_b0, ph_w1, ph_b1, ph_w2, ph_b2,
                                       h_cur, x_cur, l);
  }
  k_out<<<(BB * NN * DD + 255) / 256, 256, 0, stream>>>(x_cur, (float*)d_out);
}
